// DeformBottleneckBlock_33457795236467
// MI455X (gfx1250) — compile-verified, matrix operand order checked by reading, not run on hardware
//
#include <hip/hip_runtime.h>

// DeformBottleneckBlock for MI455X (gfx1250, wave32).
// GEMMs on v_wmma_f32_16x16x32_bf16 with hi/lo bf16 split (~fp32 accuracy).
// Operands are pre-swizzled once into fragment-ready bf16 layouts so the
// GEMM inner loop is pure b128 loads + WMMA (no per-iteration conversions).

#define HWp 4096   // 64*64 pixels per image
#define Hd  64
#define Wd  64
#define NB  2      // batch
#define NT  (HWp / 16)   // 256 n-tiles per batch

typedef __attribute__((ext_vector_type(16))) __bf16 v16bf;
typedef __attribute__((ext_vector_type(8)))  float  v8f;

__device__ __forceinline__ unsigned short f32_bf16_rn(float f) {
  unsigned u = __builtin_bit_cast(unsigned, f);
  u += 0x7FFFu + ((u >> 16) & 1u);          // round-to-nearest-even
  return (unsigned short)(u >> 16);
}
__device__ __forceinline__ float bf16_f32(unsigned short h) {
  unsigned u = ((unsigned)h) << 16;
  return __builtin_bit_cast(float, u);
}
__device__ __forceinline__ void split_u16(float f, unsigned short& hi, unsigned short& lo) {
  hi = f32_bf16_rn(f);
  lo = f32_bf16_rn(f - bf16_f32(hi));
}

// Fragment storage: one fragment = 16x32 (or 32x16) bf16 tile.
// Layout: frag*1024 + lane*32 + e  (u16 units); hi at [e], lo at [e+16].
// Per the CDNA5 16-bit WMMA layout, element e of lane L holds
//   row = (L&15), k = kt*32 + ((e>>1)>>2)*16 + (L>>4)*8 + ((e>>1)&3)*2 + (e&1).
__device__ __forceinline__ void frag_pos(int kr /*k&31*/, int col, int& lane, int& e) {
  const int i    = ((kr >> 4) << 2) | ((kr >> 1) & 3);
  const int half = (kr >> 3) & 1;
  e    = 2 * i + (kr & 1);
  lane = (half << 4) | (col & 15);
}

// ---- Weight (A) fragment conversion: A f32 [M,K] -> AF bf16 frags ----
__global__ __launch_bounds__(256)
void convert_A_frags(const float* __restrict__ A, __bf16* __restrict__ AF,
                     int M, int K)
{
  const int t = blockIdx.x * blockDim.x + threadIdx.x;
  if (t >= M * K) return;
  const int m = t / K, k = t % K;     // coalesced source reads along k
  const int Kt = K >> 5;
  int lane, e;
  frag_pos(k & 31, m, lane, e);
  const size_t frag = (size_t)(m >> 4) * Kt + (k >> 5);
  unsigned short hi, lo;
  split_u16(A[t], hi, lo);
  __bf16* p = AF + frag * 1024 + lane * 32;
  p[e]      = __builtin_bit_cast(__bf16, hi);
  p[e + 16] = __builtin_bit_cast(__bf16, lo);
}

// ---- Activation (B) fragment conversion: Bsrc f32 [batch][K,HWp] -> BF ----
__global__ __launch_bounds__(256)
void convert_B_frags(const float* __restrict__ Bsrc, __bf16* __restrict__ BF,
                     int K)
{
  const int t = blockIdx.x * blockDim.x + threadIdx.x;
  const int b = blockIdx.y;
  if (t >= K * HWp) return;
  const int k = t / HWp, n = t % HWp; // coalesced source reads along n
  const int Kt = K >> 5;
  int lane, e;
  frag_pos(k & 31, n, lane, e);
  const size_t frag = (size_t)(k >> 5) * NT + (n >> 4);
  unsigned short hi, lo;
  split_u16(Bsrc[(size_t)b * K * HWp + t], hi, lo);
  __bf16* p = BF + ((size_t)b * Kt * NT + frag) * 1024 + lane * 32;
  p[e]      = __builtin_bit_cast(__bf16, hi);
  p[e + 16] = __builtin_bit_cast(__bf16, lo);
}

// ---- WMMA GEMM: out = relu?( bn( A@B ) (+res) ), one wave per 16x16 tile ----
__global__ __launch_bounds__(128)
void wmma_gemm_bn(const __bf16* __restrict__ AF, const __bf16* __restrict__ BF,
                  int K, int M,
                  const float* __restrict__ gam, const float* __restrict__ bet,
                  const float* __restrict__ mu,  const float* __restrict__ var,
                  const float* __restrict__ res,      // nullable [batch][M,HWp]
                  float* __restrict__ out, int relu)
{
  const int lane = threadIdx.x & 31;
  const int wave = threadIdx.x >> 5;
  const int half = lane >> 4;
  const int lq   = lane & 15;
  const int mt   = blockIdx.y * 4 + wave;
  const int nt   = blockIdx.x;
  const int b    = blockIdx.z;
  const int Kt   = K >> 5;
  if (mt * 16 >= M) return;           // wave-uniform (EXEC stays all-1s)

  const __bf16* Af = AF + ((size_t)mt * Kt) * 1024 + lane * 32;
  const __bf16* Bf = BF + ((size_t)b * Kt * NT + nt) * 1024 + lane * 32;

  v8f acc = {};
  for (int kt = 0; kt < Kt; kt += 2) {
#pragma unroll
    for (int u = 0; u < 2; ++u) {
      __builtin_prefetch(Bf + (size_t)NT * 1024, 0, 1);  // global_prefetch_b8
      const v16bf ahi = *(const v16bf*)(Af);
      const v16bf alo = *(const v16bf*)(Af + 16);
      const v16bf bhi = *(const v16bf*)(Bf);
      const v16bf blo = *(const v16bf*)(Bf + 16);
      // fp32-accurate product via bf16 split: hi*hi + hi*lo + lo*hi
      acc = __builtin_amdgcn_wmma_f32_16x16x32_bf16(false, ahi, false, bhi,
                                                    (short)0, acc, false, false);
      acc = __builtin_amdgcn_wmma_f32_16x16x32_bf16(false, ahi, false, blo,
                                                    (short)0, acc, false, false);
      acc = __builtin_amdgcn_wmma_f32_16x16x32_bf16(false, alo, false, bhi,
                                                    (short)0, acc, false, false);
      Af += 1024;
      Bf += (size_t)NT * 1024;
    }
  }

  float* Ob = out + (size_t)b * M * HWp;
  const float* Rb = res ? res + (size_t)b * M * HWp : nullptr;
  const int n = nt * 16 + lq;
#pragma unroll
  for (int r = 0; r < 8; ++r) {
    const int m = mt * 16 + r + half * 8;   // D layout: VGPR r -> M = r + half*8
    const float scale = gam[m] * rsqrtf(var[m] + 1e-5f);
    const float shift = bet[m] - mu[m] * scale;
    float v = acc[r] * scale + shift;
    if (Rb) v += Rb[(size_t)m * HWp + n];
    if (relu) v = fmaxf(v, 0.f);
    Ob[(size_t)m * HWp + n] = v;
  }
}

// ---- 3x3 offset conv, 18 output channels (VALU; too skinny for WMMA) ----
__global__ __launch_bounds__(256)
void offset_conv3x3(const float* __restrict__ C1, const float* __restrict__ w_off,
                    const float* __restrict__ b_off, float* __restrict__ offs)
{
  const int t = blockIdx.x * blockDim.x + threadIdx.x;
  if (t >= NB * 18 * HWp) return;
  const int p  = t & (HWp - 1);
  const int oc = (t >> 12) % 18;
  const int b  = t / (18 * HWp);
  const int y = p >> 6, x = p & 63;
  const float* Cb = C1 + (size_t)b * 256 * HWp;
  float acc = b_off[oc];
#pragma unroll
  for (int ky = 0; ky < 3; ++ky) {
    const int yy = y + ky - 1;
    if (yy < 0 || yy >= Hd) continue;
#pragma unroll
    for (int kx = 0; kx < 3; ++kx) {
      const int xx = x + kx - 1;
      if (xx < 0 || xx >= Wd) continue;
      const int idx = yy * Wd + xx;
      const float* wp = w_off + (size_t)oc * 256 * 9 + ky * 3 + kx;
      for (int c = 0; c < 256; ++c)
        acc += wp[c * 9] * Cb[(size_t)c * HWp + idx];
    }
  }
  offs[(size_t)b * 18 * HWp + (size_t)oc * HWp + p] = acc;
}

// ---- Bilinear deformable sampling, writing directly into B-fragments ----
__global__ __launch_bounds__(256)
void deform_sample_frag(const float* __restrict__ C1, const float* __restrict__ offs,
                        __bf16* __restrict__ SF)
{
  const int t = blockIdx.x * blockDim.x + threadIdx.x;
  if (t >= NB * 9 * HWp) return;
  const int p  = t & (HWp - 1);
  const int kq = (t >> 12) % 9;
  const int b  = t / (9 * HWp);
  const int y = p >> 6, x = p & 63;

  const float* ob = offs + (size_t)b * 18 * HWp;
  const float dy = ob[(size_t)(2 * kq) * HWp + p];
  const float dx = ob[(size_t)(2 * kq + 1) * HWp + p];
  const float py = (float)y + (float)(kq / 3 - 1) + dy;
  const float px = (float)x + (float)(kq % 3 - 1) + dx;

  const float y0f = floorf(py), x0f = floorf(px);
  const float wy1 = py - y0f,   wx1 = px - x0f;
  const int y0 = (int)y0f, x0 = (int)x0f;
  const int y1 = y0 + 1,   x1 = x0 + 1;

  auto inH = [](int v) { return v >= 0 && v < Hd; };
  auto inW = [](int v) { return v >= 0 && v < Wd; };
  float w00 = (1.f - wy1) * (1.f - wx1) * ((inH(y0) && inW(x0)) ? 1.f : 0.f);
  float w01 = (1.f - wy1) * wx1         * ((inH(y0) && inW(x1)) ? 1.f : 0.f);
  float w10 = wy1 * (1.f - wx1)         * ((inH(y1) && inW(x0)) ? 1.f : 0.f);
  float w11 = wy1 * wx1                 * ((inH(y1) && inW(x1)) ? 1.f : 0.f);
  auto cl = [](int v, int hi) { return v < 0 ? 0 : (v > hi ? hi : v); };
  const int i00 = cl(y0, Hd - 1) * Wd + cl(x0, Wd - 1);
  const int i01 = cl(y0, Hd - 1) * Wd + cl(x1, Wd - 1);
  const int i10 = cl(y1, Hd - 1) * Wd + cl(x0, Wd - 1);
  const int i11 = cl(y1, Hd - 1) * Wd + cl(x1, Wd - 1);

  const float* Cb = C1 + (size_t)b * 256 * HWp;
  const int Kt = 2304 >> 5;   // 72
  __bf16* SFb = SF + (size_t)b * Kt * NT * 1024;
  const int nt = p >> 4;
  for (int c = 0; c < 256; ++c) {
    const float* cp = Cb + (size_t)c * HWp;
    const float v = w00 * cp[i00] + w01 * cp[i01] + w10 * cp[i10] + w11 * cp[i11];
    const int k = c * 9 + kq;
    int lane, e;
    frag_pos(k & 31, p, lane, e);
    unsigned short hi, lo;
    split_u16(v, hi, lo);
    __bf16* q = SFb + ((size_t)(k >> 5) * NT + nt) * 1024 + lane * 32;
    q[e]      = __builtin_bit_cast(__bf16, hi);
    q[e + 16] = __builtin_bit_cast(__bf16, lo);
  }
}

extern "C" void kernel_launch(void* const* d_in, const int* in_sizes, int n_in,
                              void* d_out, int out_size, void* d_ws, size_t ws_size,
                              hipStream_t stream) {
  (void)in_sizes; (void)n_in; (void)out_size; (void)ws_size;
  const float* x     = (const float*)d_in[0];
  const float* w1    = (const float*)d_in[1];
  const float* g1    = (const float*)d_in[2];
  const float* b1    = (const float*)d_in[3];
  const float* m1    = (const float*)d_in[4];
  const float* v1    = (const float*)d_in[5];
  const float* w_off = (const float*)d_in[6];
  const float* b_off = (const float*)d_in[7];
  const float* w2    = (const float*)d_in[8];
  const float* g2    = (const float*)d_in[9];
  const float* b2    = (const float*)d_in[10];
  const float* m2    = (const float*)d_in[11];
  const float* v2    = (const float*)d_in[12];
  const float* w3    = (const float*)d_in[13];
  const float* g3    = (const float*)d_in[14];
  const float* b3    = (const float*)d_in[15];
  const float* m3    = (const float*)d_in[16];
  const float* v3    = (const float*)d_in[17];
  float* out = (float*)d_out;

  // ---- workspace layout (~139 MB total) ----
  char* wsp = (char*)d_ws;
  float*  C1  = (float*)wsp;  wsp += (size_t)NB * 256 * HWp * 4;   // plain f32
  float*  OFF = (float*)wsp;  wsp += (size_t)NB * 18  * HWp * 4;
  float*  C2  = (float*)wsp;  wsp += (size_t)NB * 256 * HWp * 4;
  __bf16* w1F = (__bf16*)wsp; wsp += (size_t)256  * 1024 * 4;      // hi+lo u16
  __bf16* w2F = (__bf16*)wsp; wsp += (size_t)256  * 2304 * 4;
  __bf16* w3F = (__bf16*)wsp; wsp += (size_t)1024 * 256  * 4;
  __bf16* xF  = (__bf16*)wsp; wsp += (size_t)NB * 1024 * HWp * 4;
  __bf16* SF  = (__bf16*)wsp; wsp += (size_t)NB * 2304 * HWp * 4;
  __bf16* C2F = (__bf16*)wsp;

  // ---- operand pre-swizzle (one-pass, bandwidth-bound, L2-resident) ----
  convert_A_frags<<<(256 * 1024 + 255) / 256, 256, 0, stream>>>(w1, w1F, 256, 1024);
  convert_A_frags<<<(256 * 2304 + 255) / 256, 256, 0, stream>>>(w2, w2F, 256, 2304);
  convert_A_frags<<<(1024 * 256 + 255) / 256, 256, 0, stream>>>(w3, w3F, 1024, 256);
  convert_B_frags<<<dim3((1024 * HWp + 255) / 256, NB), 256, 0, stream>>>(x, xF, 1024);

  // 1) conv1x1 #1 + BN + ReLU  (M=256, K=1024)
  wmma_gemm_bn<<<dim3(NT, 4, NB), 128, 0, stream>>>(
      w1F, xF, 1024, 256, g1, b1, m1, v1, nullptr, C1, 1);
  // 2) offset conv 3x3 (18 channels)
  offset_conv3x3<<<(NB * 18 * HWp + 255) / 256, 256, 0, stream>>>(
      C1, w_off, b_off, OFF);
  // 3) bilinear deformable sampling -> SF fragments [K=2304]
  deform_sample_frag<<<(NB * 9 * HWp + 255) / 256, 256, 0, stream>>>(C1, OFF, SF);
  // 4) deform conv GEMM + BN + ReLU  (M=256, K=2304)
  wmma_gemm_bn<<<dim3(NT, 4, NB), 128, 0, stream>>>(
      w2F, SF, 2304, 256, g2, b2, m2, v2, nullptr, C2, 1);
  // 5) re-swizzle C2 then conv1x1 #2 + BN + residual(x) + ReLU  (M=1024, K=256)
  convert_B_frags<<<dim3((256 * HWp + 255) / 256, NB), 256, 0, stream>>>(C2, C2F, 256);
  wmma_gemm_bn<<<dim3(NT, 16, NB), 128, 0, stream>>>(
      w3F, C2F, 256, 1024, g3, b3, m3, v3, x, out, 1);
}
